// TokenSpikingTransformer_88983132438984
// MI455X (gfx1250) — compile-verified, matrix operand order checked by reading, not run on hardware
//
#include <hip/hip_runtime.h>
#include <hip/hip_bf16.h>

typedef __attribute__((ext_vector_type(16))) _Float16 v16h;
typedef __attribute__((ext_vector_type(8)))  float    v8f;
typedef _Float16 half_t;

#define TDIM  4
#define BB    16
#define CC    768
#define NN    256
#define COLS  4096           // BB * NN
#define CHD   3072
#define HEADS 12
#define HDIM  64

// ---------------------------------------------------------------------------
// BN parameter prep: scale = g * rsqrt(var+eps), shift = beta + (bias-mean)*scale
// ---------------------------------------------------------------------------
__global__ __launch_bounds__(256)
void bn_prep_kernel(const float* __restrict__ bn, const float* __restrict__ bias,
                    float* __restrict__ scale, float* __restrict__ shift, int C) {
    int i = blockIdx.x * 256 + threadIdx.x;
    if (i >= C) return;
    float g = bn[i], b = bn[C + i], m = bn[2 * C + i], v = bn[3 * C + i];
    float sc = g * rsqrtf(v + 1e-5f);
    float bi = bias ? bias[i] : 0.0f;
    scale[i] = sc;
    shift[i] = b + (bi - m) * sc;
}

// ---------------------------------------------------------------------------
// fp32 -> fp16 weight cast (same row-major [M,K] layout = WMMA A operand feed)
// ---------------------------------------------------------------------------
__global__ __launch_bounds__(256)
void cast_f32_f16_kernel(const float* __restrict__ in, half_t* __restrict__ out, int n) {
    int i = blockIdx.x * 256 + threadIdx.x;
    if (i < n) out[i] = (half_t)in[i];
}

// ---------------------------------------------------------------------------
// x [T,B,C,N] fp32  ->  xt [T, COLS, C] fp16   (col = b*N + n)
// Output-channel-contiguous layout == WMMA B operand layout for next GEMM.
// ---------------------------------------------------------------------------
__global__ __launch_bounds__(256)
void transpose_cast_x_kernel(const float* __restrict__ x, half_t* __restrict__ xt) {
    size_t i = (size_t)blockIdx.x * 256 + threadIdx.x;     // over T*COLS*C
    int c = (int)(i % CC);
    size_t rest = i / CC;
    int col = (int)(rest % COLS);
    int t = (int)(rest / COLS);
    int b = col >> 8, n = col & 255;
    xt[i] = (half_t)x[(((size_t)t * BB + b) * CC + c) * NN + n];
}

// ---------------------------------------------------------------------------
// Fused GEMM + BN + multi-step LIF.
//   A: [M,K] f16 weights (row-major)
//   B: [T][COLS][K] f16 activations
//   Per wave: 32(M) x 16(col) output tile, 4 timestep accumulators sharing the
//   A fragment; LIF scan over T done in registers in the epilogue.
// mode 0: write spikes -> out_spk [T][COLS][M] f16
// mode 1: x2 = resid_in(x, [T,B,C,N] f32) + spike; write x2_f32 & x2_f16 [T][COLS][M]
// mode 2: final = x2_f32 + spike; write fp32 [T,B,C,H*W] layout (d_out)
// ---------------------------------------------------------------------------
__global__ __launch_bounds__(256)
void gemm_bn_lif_kernel(const half_t* __restrict__ A,
                        const half_t* __restrict__ B,
                        const float* __restrict__ scale,
                        const float* __restrict__ shift,
                        half_t* __restrict__ out_spk,
                        const float* __restrict__ resid_in,
                        float* __restrict__ x2_f32,
                        half_t* __restrict__ x2_f16,
                        float* __restrict__ final_out,
                        int M, int K, int mode) {
    const int lane = threadIdx.x & 31;
    const int wave = threadIdx.x >> 5;
    const int wm = wave & 1;            // 2 waves along M
    const int wn = wave >> 1;           // 4 waves along cols
    const int m0  = blockIdx.x * 64 + wm * 32;
    const int col = blockIdx.y * 64 + wn * 16 + (lane & 15);
    const int kh  = (lane >> 4) * 16;   // K-half select (contiguous k order, same for A and B)

    const half_t* arow0 = A + (size_t)(m0 + (lane & 15)) * K + kh;
    const half_t* arow1 = arow0 + (size_t)16 * K;
    const half_t* bcol  = B + (size_t)col * K + kh;
    const size_t  bstr  = (size_t)COLS * K;   // per-timestep stride

    v8f acc[2][TDIM];
    #pragma unroll
    for (int s = 0; s < 2; ++s)
        #pragma unroll
        for (int t = 0; t < TDIM; ++t)
            acc[s][t] = (v8f)0.0f;

    for (int k0 = 0; k0 < K; k0 += 32) {
        v16h a0 = *(const v16h*)(arow0 + k0);
        v16h a1 = *(const v16h*)(arow1 + k0);
        v16h b0 = *(const v16h*)(bcol + k0);
        v16h b1 = *(const v16h*)(bcol + bstr + k0);
        v16h b2 = *(const v16h*)(bcol + 2 * bstr + k0);
        v16h b3 = *(const v16h*)(bcol + 3 * bstr + k0);
        acc[0][0] = __builtin_amdgcn_wmma_f32_16x16x32_f16(false, a0, false, b0, (short)0, acc[0][0], false, false);
        acc[1][0] = __builtin_amdgcn_wmma_f32_16x16x32_f16(false, a1, false, b0, (short)0, acc[1][0], false, false);
        acc[0][1] = __builtin_amdgcn_wmma_f32_16x16x32_f16(false, a0, false, b1, (short)0, acc[0][1], false, false);
        acc[1][1] = __builtin_amdgcn_wmma_f32_16x16x32_f16(false, a1, false, b1, (short)0, acc[1][1], false, false);
        acc[0][2] = __builtin_amdgcn_wmma_f32_16x16x32_f16(false, a0, false, b2, (short)0, acc[0][2], false, false);
        acc[1][2] = __builtin_amdgcn_wmma_f32_16x16x32_f16(false, a1, false, b2, (short)0, acc[1][2], false, false);
        acc[0][3] = __builtin_amdgcn_wmma_f32_16x16x32_f16(false, a0, false, b3, (short)0, acc[0][3], false, false);
        acc[1][3] = __builtin_amdgcn_wmma_f32_16x16x32_f16(false, a1, false, b3, (short)0, acc[1][3], false, false);
    }

    // Epilogue: BN + 4-step LIF scan per output element, then mode-specific write.
    // D layout: lane<16 -> (M=r, N=lane); lane>=16 -> (M=r+8, N=lane-16).
    const int rowhi = (lane >> 4) << 3;
    #pragma unroll
    for (int s = 0; s < 2; ++s) {
        #pragma unroll
        for (int r = 0; r < 8; ++r) {
            const int row = m0 + s * 16 + r + rowhi;
            const float sc = scale[row];
            const float sh = shift[row];
            float v = 0.0f;
            float sp[TDIM];
            #pragma unroll
            for (int t = 0; t < TDIM; ++t) {
                float y = acc[s][t][r] * sc + sh;
                v = (v + y) * 0.5f;                    // v += (y - v)/tau, tau = 2
                float fire = (v >= 1.0f) ? 1.0f : 0.0f;
                v *= (1.0f - fire);                    // hard reset
                sp[t] = fire;
            }
            #pragma unroll
            for (int t = 0; t < TDIM; ++t) {
                const size_t oidx = ((size_t)t * COLS + col) * M + row;
                if (mode == 0) {
                    out_spk[oidx] = (half_t)sp[t];
                } else if (mode == 1) {
                    const int b = col >> 8, n = col & 255;
                    const size_t xi = (((size_t)t * BB + b) * CC + row) * NN + n;
                    const float xv = resid_in[xi] + sp[t];
                    x2_f32[oidx] = xv;
                    x2_f16[oidx] = (half_t)xv;
                } else {
                    const int b = col >> 8, n = col & 255;
                    const size_t xo = (((size_t)t * BB + b) * CC + row) * NN + n;
                    final_out[xo] = x2_f32[oidx] + sp[t];
                }
            }
        }
    }
}

// ---------------------------------------------------------------------------
// Attention: per (col, head): y_t = sum_d q[t][col][h*64+d]; attn = LIF(y);
// a[t][col][h*64+d] = attn_t * k[t][col][h*64+d]
// ---------------------------------------------------------------------------
__global__ __launch_bounds__(256)
void attn_lif_kernel(const half_t* __restrict__ q_s,
                     const half_t* __restrict__ k_s,
                     half_t* __restrict__ a_out) {
    int tid = blockIdx.x * 256 + threadIdx.x;
    if (tid >= COLS * HEADS) return;
    const int h = tid % HEADS;
    const int col = tid / HEADS;
    const size_t base = (size_t)col * CC + h * HDIM;
    const size_t tstr = (size_t)COLS * CC;

    float v = 0.0f;
    float sp[TDIM];
    #pragma unroll
    for (int t = 0; t < TDIM; ++t) {
        const v16h* qp = (const v16h*)(q_s + t * tstr + base);
        float sum = 0.0f;
        #pragma unroll
        for (int i = 0; i < 4; ++i) {
            v16h qv = qp[i];
            #pragma unroll
            for (int j = 0; j < 16; ++j) sum += (float)qv[j];
        }
        v = (v + sum) * 0.5f;
        float fire = (v >= 1.0f) ? 1.0f : 0.0f;
        v *= (1.0f - fire);
        sp[t] = fire;
    }
    #pragma unroll
    for (int t = 0; t < TDIM; ++t) {
        const v16h* kp = (const v16h*)(k_s + t * tstr + base);
        v16h* ap = (v16h*)(a_out + t * tstr + base);
        half_t g = (half_t)sp[t];
        #pragma unroll
        for (int i = 0; i < 4; ++i) ap[i] = kp[i] * g;
    }
}

// ---------------------------------------------------------------------------
extern "C" void kernel_launch(void* const* d_in, const int* in_sizes, int n_in,
                              void* d_out, int out_size, void* d_ws, size_t ws_size,
                              hipStream_t stream) {
    const float* x    = (const float*)d_in[0];
    const float* w_q  = (const float*)d_in[1];
    const float* bn_q = (const float*)d_in[2];
    const float* w_k  = (const float*)d_in[3];
    const float* bn_k = (const float*)d_in[4];
    const float* w_p  = (const float*)d_in[5];
    const float* b_p  = (const float*)d_in[6];
    const float* bn_p = (const float*)d_in[7];
    const float* w1   = (const float*)d_in[8];
    const float* b1   = (const float*)d_in[9];
    const float* bn1  = (const float*)d_in[10];
    const float* w2   = (const float*)d_in[11];
    const float* b2   = (const float*)d_in[12];
    const float* bn2  = (const float*)d_in[13];
    float* out = (float*)d_out;

    // --- workspace carve-out ---
    char* ws = (char*)d_ws;
    size_t off = 0;
    auto take = [&](size_t bytes) -> void* {
        off = (off + 255) & ~(size_t)255;
        void* p = ws + off;
        off += bytes;
        return p;
    };
    const size_t act_h = (size_t)TDIM * COLS * CC * sizeof(half_t);   // 25.2 MB
    half_t* xt     = (half_t*)take(act_h);
    half_t* wq_h   = (half_t*)take((size_t)CC * CC * sizeof(half_t));
    half_t* wk_h   = (half_t*)take((size_t)CC * CC * sizeof(half_t));
    half_t* wp_h   = (half_t*)take((size_t)CC * CC * sizeof(half_t));
    half_t* w1_h   = (half_t*)take((size_t)CHD * CC * sizeof(half_t));
    half_t* w2_h   = (half_t*)take((size_t)CC * CHD * sizeof(half_t));
    half_t* q_s    = (half_t*)take(act_h);
    half_t* k_s    = (half_t*)take(act_h);
    half_t* a_h    = (half_t*)take(act_h);
    half_t* x2_h   = (half_t*)take(act_h);
    float*  x2_f   = (float*)take((size_t)TDIM * COLS * CC * sizeof(float));
    half_t* s1     = (half_t*)take((size_t)TDIM * COLS * CHD * sizeof(half_t));
    float*  sc_q   = (float*)take(CC * 4);   float* sh_q = (float*)take(CC * 4);
    float*  sc_k   = (float*)take(CC * 4);   float* sh_k = (float*)take(CC * 4);
    float*  sc_p   = (float*)take(CC * 4);   float* sh_p = (float*)take(CC * 4);
    float*  sc_1   = (float*)take(CHD * 4);  float* sh_1 = (float*)take(CHD * 4);
    float*  sc_2   = (float*)take(CC * 4);   float* sh_2 = (float*)take(CC * 4);
    (void)ws_size; (void)in_sizes; (void)n_in; (void)out_size;

    // --- BN param prep (bias folded into shift) ---
    bn_prep_kernel<<<(CC + 255) / 256, 256, 0, stream>>>(bn_q, nullptr, sc_q, sh_q, CC);
    bn_prep_kernel<<<(CC + 255) / 256, 256, 0, stream>>>(bn_k, nullptr, sc_k, sh_k, CC);
    bn_prep_kernel<<<(CC + 255) / 256, 256, 0, stream>>>(bn_p, b_p, sc_p, sh_p, CC);
    bn_prep_kernel<<<(CHD + 255) / 256, 256, 0, stream>>>(bn1, b1, sc_1, sh_1, CHD);
    bn_prep_kernel<<<(CC + 255) / 256, 256, 0, stream>>>(bn2, b2, sc_2, sh_2, CC);

    // --- weight casts ---
    cast_f32_f16_kernel<<<(CC * CC + 255) / 256, 256, 0, stream>>>(w_q, wq_h, CC * CC);
    cast_f32_f16_kernel<<<(CC * CC + 255) / 256, 256, 0, stream>>>(w_k, wk_h, CC * CC);
    cast_f32_f16_kernel<<<(CC * CC + 255) / 256, 256, 0, stream>>>(w_p, wp_h, CC * CC);
    cast_f32_f16_kernel<<<(CHD * CC + 255) / 256, 256, 0, stream>>>(w1, w1_h, CHD * CC);
    cast_f32_f16_kernel<<<(CC * CHD + 255) / 256, 256, 0, stream>>>(w2, w2_h, CC * CHD);

    // --- x -> [t][col][c] f16 ---
    {
        size_t total = (size_t)TDIM * COLS * CC;
        transpose_cast_x_kernel<<<(unsigned)(total / 256), 256, 0, stream>>>(x, xt);
    }

    dim3 blk(256);
    dim3 grid_c(CC / 64, COLS / 64);     // M=768
    dim3 grid_h(CHD / 64, COLS / 64);    // M=3072

    // Q = LIF(BN(w_q @ x))
    gemm_bn_lif_kernel<<<grid_c, blk, 0, stream>>>(wq_h, xt, sc_q, sh_q, q_s,
                                                   nullptr, nullptr, nullptr, nullptr,
                                                   CC, CC, 0);
    // K = LIF(BN(w_k @ x))
    gemm_bn_lif_kernel<<<grid_c, blk, 0, stream>>>(wk_h, xt, sc_k, sh_k, k_s,
                                                   nullptr, nullptr, nullptr, nullptr,
                                                   CC, CC, 0);
    // attn gating: a = LIF(sum_d q) * k
    attn_lif_kernel<<<(COLS * HEADS + 255) / 256, 256, 0, stream>>>(q_s, k_s, a_h);

    // x2 = x + LIF(BN(w_p @ a + b_p))
    gemm_bn_lif_kernel<<<grid_c, blk, 0, stream>>>(wp_h, a_h, sc_p, sh_p, nullptr,
                                                   x, x2_f, x2_h, nullptr,
                                                   CC, CC, 1);
    // s1 = LIF(BN(w1 @ x2 + b1))
    gemm_bn_lif_kernel<<<grid_h, blk, 0, stream>>>(w1_h, x2_h, sc_1, sh_1, s1,
                                                   nullptr, nullptr, nullptr, nullptr,
                                                   CHD, CC, 0);
    // out = x2 + LIF(BN(w2 @ s1 + b2))
    gemm_bn_lif_kernel<<<grid_c, blk, 0, stream>>>(w2_h, s1, sc_2, sh_2, nullptr,
                                                   nullptr, x2_f, nullptr, out,
                                                   CC, CHD, 2);
}
